// Head_33663953666991
// MI455X (gfx1250) — matrix operand order repaired and checked by reading, not compiled, not run on hardware
//
#include <hip/hip_runtime.h>
#include <hip/hip_bf16.h>

typedef _Float16 half_t;
typedef __attribute__((ext_vector_type(16))) _Float16 v16h;
typedef __attribute__((ext_vector_type(8)))  _Float16 v8h;
typedef __attribute__((ext_vector_type(8)))  float    v8f;
typedef __attribute__((ext_vector_type(4)))  float    v4f;

#define BATCH 4
#define T     4096
#define CDIM  512
#define HDIM  64
#define NROW  (BATCH*T)        // 16384 total rows

// ---------------------------------------------------------------------------
// Fragment helpers (layouts per CDNA5 ISA 7.12.2, wave32)
// A (16x32 f16): lane L holds row M=L&15; halfs j=0..7 -> K=base+j,
//                j=8..15 -> K=base+16+(j-8), base = 8*(L>>4).
// B (32x16 f16): lane L holds col N=L&15; same K mapping as A (base = 8*(L>>4)).
// C/D (16x16 f32): element r of v8f -> (M = r + 8*(L>>4), N = L&15).
// ---------------------------------------------------------------------------

__device__ inline v16h load_b_frag_f16(const half_t* p) {
  // p already points at (col N)*ld + 8*(lane>>4); chunks [0..7] and [16..23]
  v8h b0 = *(const v8h*)(p);
  v8h b1 = *(const v8h*)(p + 16);
  return __builtin_shufflevector(b0, b1, 0,1,2,3,4,5,6,7,8,9,10,11,12,13,14,15);
}

__device__ inline v16h load_a_frag_f16(const half_t* p) {
  // p points at row*ld + kbase + 8*(lane>>4); chunks [0..7] and [16..23]
  v8h a0 = *(const v8h*)(p);
  v8h a1 = *(const v8h*)(p + 16);
  return __builtin_shufflevector(a0, a1, 0,1,2,3,4,5,6,7,8,9,10,11,12,13,14,15);
}

__device__ inline v8f wmma_f16(v16h a, v16h b, v8f c) {
  return __builtin_amdgcn_wmma_f32_16x16x32_f16(false, a, false, b, (short)0, c,
                                                false, false);
}

// ---------------------------------------------------------------------------
// Kernel 0: Wt[m][n][k] = (f16) W_m[k][n]   (3 x 64 x 512)
// ---------------------------------------------------------------------------
__global__ __launch_bounds__(256) void prep_weights(const float* __restrict__ Wk,
                                                    const float* __restrict__ Wq,
                                                    const float* __restrict__ Wv,
                                                    half_t* __restrict__ Wt) {
  int idx = blockIdx.x * 256 + threadIdx.x;
  if (idx >= 3 * HDIM * CDIM) return;
  int m   = idx / (HDIM * CDIM);
  int rem = idx % (HDIM * CDIM);
  int n   = rem / CDIM;
  int k   = rem % CDIM;
  const float* W = (m == 0) ? Wk : (m == 1) ? Wq : Wv;
  Wt[idx] = (half_t)W[k * HDIM + n];
}

// ---------------------------------------------------------------------------
// Kernel 1: projections. One wave (32 thr) per 16-row tile.
// K/Q stored row-major f16 [row][64]; V stored transposed Vt[b][h][t].
// ---------------------------------------------------------------------------
__global__ __launch_bounds__(32) void proj_kernel(const float* __restrict__ x,
                                                  const half_t* __restrict__ Wt,
                                                  const float* __restrict__ bk,
                                                  const float* __restrict__ bq,
                                                  const float* __restrict__ bv,
                                                  half_t* __restrict__ Kf,
                                                  half_t* __restrict__ Qf,
                                                  half_t* __restrict__ Vt) {
  const int l  = threadIdx.x;
  const int lm = l & 15;
  const int lh = l >> 4;
  const size_t row0 = (size_t)blockIdx.x * 16;

  v8f acc[3][4];
#pragma unroll
  for (int m = 0; m < 3; ++m)
#pragma unroll
    for (int nt = 0; nt < 4; ++nt)
#pragma unroll
      for (int r = 0; r < 8; ++r) acc[m][nt][r] = 0.0f;

  const float* arow = x + (row0 + (size_t)lm) * CDIM + 8 * lh;

  for (int k0 = 0; k0 < CDIM; k0 += 32) {
    // A fragment: f32 -> f16 on the fly (two aligned 16B pairs)
    const float* r = arow + k0;
    v4f f0 = *(const v4f*)(r);
    v4f f1 = *(const v4f*)(r + 4);
    v4f f2 = *(const v4f*)(r + 16);
    v4f f3 = *(const v4f*)(r + 20);
    v16h a;
#pragma unroll
    for (int j = 0; j < 4; ++j) {
      a[j]      = (_Float16)f0[j];
      a[4 + j]  = (_Float16)f1[j];
      a[8 + j]  = (_Float16)f2[j];
      a[12 + j] = (_Float16)f3[j];
    }
#pragma unroll
    for (int m = 0; m < 3; ++m) {
      const half_t* wb = Wt + (size_t)m * (HDIM * CDIM);
#pragma unroll
      for (int nt = 0; nt < 4; ++nt) {
        const half_t* p = wb + (size_t)(nt * 16 + lm) * CDIM + k0 + 8 * lh;
        v16h b = load_b_frag_f16(p);
        acc[m][nt] = wmma_f16(a, b, acc[m][nt]);
      }
    }
  }

  const float* biases[3] = {bk, bq, bv};
#pragma unroll
  for (int m = 0; m < 3; ++m) {
#pragma unroll
    for (int nt = 0; nt < 4; ++nt) {
      const int h = nt * 16 + lm;
      const float bias = biases[m][h];
#pragma unroll
      for (int r = 0; r < 8; ++r) {
        const float val = acc[m][nt][r] + bias;
        const size_t row = row0 + r + 8 * lh;
        if (m == 0) {
          Kf[row * HDIM + h] = (half_t)val;
        } else if (m == 1) {
          Qf[row * HDIM + h] = (half_t)val;
        } else {
          const size_t b = row >> 12;       // row / T
          const size_t t = row & (T - 1);   // row % T
          Vt[(b * HDIM + h) * T + t] = (half_t)val;
        }
      }
    }
  }
}

// ---------------------------------------------------------------------------
// Kernel 2: flash attention. out[i,:] = softmax_j(k[i]·q[j]·scale) @ v
// One wave per 16-row tile of i. Online softmax over j in tiles of 32.
// ---------------------------------------------------------------------------
__global__ __launch_bounds__(32) void attn_kernel(const half_t* __restrict__ Kf,
                                                  const half_t* __restrict__ Qf,
                                                  const half_t* __restrict__ Vt,
                                                  float* __restrict__ out) {
  __shared__ float  Sl[16][32];
  __shared__ half_t Pl[16][32];
  __shared__ float  rowscale[16];
  __shared__ float  lrow[16];

  const int l  = threadIdx.x;
  const int lm = l & 15;
  const int lh = l >> 4;
  const size_t row0 = (size_t)blockIdx.x * 16;
  const int bidx = (int)(row0 >> 12);                 // batch
  const half_t* Qb = Qf + (size_t)bidx * T * HDIM;    // [t][64]
  const half_t* Vb = Vt + (size_t)bidx * HDIM * T;    // [h][t]
  const float scale = 0.04419417382415922f;           // 512^-0.5

  // K-row A fragments (invariant over j loop): h chunks 0..31 and 32..63
  v16h kfrag[2];
#pragma unroll
  for (int c = 0; c < 2; ++c)
    kfrag[c] = load_a_frag_f16(Kf + (row0 + (size_t)lm) * HDIM + 32 * c + 8 * lh);

  v8f acc[4];
#pragma unroll
  for (int f = 0; f < 4; ++f)
#pragma unroll
    for (int r = 0; r < 8; ++r) acc[f][r] = 0.0f;

  float m_i = -1.0e30f;
  float l_i = 0.0f;

  for (int jt = 0; jt < T; jt += 32) {
    // ---- S = K · Qᵀ  (16 x 32 tile, two 16-col halves, K-dim = 64) ----
    v8f s[2];
#pragma unroll
    for (int co = 0; co < 2; ++co) {
      v8f z;
#pragma unroll
      for (int r = 0; r < 8; ++r) z[r] = 0.0f;
      const half_t* q0 = Qb + (size_t)(jt + co * 16 + lm) * HDIM + 8 * lh;
      s[co] = wmma_f16(kfrag[0], load_b_frag_f16(q0), z);
      const half_t* q1 = q0 + 32;
      s[co] = wmma_f16(kfrag[1], load_b_frag_f16(q1), s[co]);
    }

    // ---- stage scaled S to LDS (C layout -> row-major) ----
#pragma unroll
    for (int r = 0; r < 8; ++r) {
      Sl[r + 8 * lh][lm]      = s[0][r] * scale;
      Sl[r + 8 * lh][16 + lm] = s[1][r] * scale;
    }
    __syncthreads();

    // ---- online softmax row pass: lanes 0..15 own one row each ----
    if (l < 16) {
      float mnew = m_i;
#pragma unroll
      for (int j = 0; j < 32; ++j) mnew = fmaxf(mnew, Sl[l][j]);
      const float corr = __expf(m_i - mnew);
      float sum = 0.0f;
#pragma unroll
      for (int j = 0; j < 32; ++j) {
        const float e = __expf(Sl[l][j] - mnew);
        Pl[l][j] = (half_t)e;
        sum += e;
      }
      l_i = l_i * corr + sum;
      m_i = mnew;
      rowscale[l] = corr;
    }
    __syncthreads();

    // ---- rescale output accumulators (per-row broadcast from LDS) ----
    float cs[8];
#pragma unroll
    for (int r = 0; r < 8; ++r) cs[r] = rowscale[r + 8 * lh];
#pragma unroll
    for (int f = 0; f < 4; ++f)
#pragma unroll
      for (int r = 0; r < 8; ++r) acc[f][r] *= cs[r];

    // ---- P A-fragment from LDS (does the lane shuffle for us) ----
    v16h pa = load_a_frag_f16(&Pl[lm][8 * lh]);

    // ---- out += P · V  (V transposed => contiguous B fragments) ----
#pragma unroll
    for (int hn = 0; hn < 4; ++hn) {
      const half_t* vp = Vb + (size_t)(hn * 16 + lm) * T + jt + 8 * lh;
      acc[hn] = wmma_f16(pa, load_b_frag_f16(vp), acc[hn]);
    }
    __syncthreads();
  }

  // ---- normalize and store f32 output ----
  if (l < 16) lrow[l] = l_i;
  __syncthreads();
  float inv[8];
#pragma unroll
  for (int r = 0; r < 8; ++r) inv[r] = 1.0f / lrow[r + 8 * lh];
#pragma unroll
  for (int hn = 0; hn < 4; ++hn)
#pragma unroll
    for (int r = 0; r < 8; ++r)
      out[(row0 + r + 8 * lh) * HDIM + hn * 16 + lm] = acc[hn][r] * inv[r];
}

// ---------------------------------------------------------------------------
extern "C" void kernel_launch(void* const* d_in, const int* in_sizes, int n_in,
                              void* d_out, int out_size, void* d_ws, size_t ws_size,
                              hipStream_t stream) {
  const float* x  = (const float*)d_in[0];
  const float* Wk = (const float*)d_in[1];
  const float* bk = (const float*)d_in[2];
  const float* Wq = (const float*)d_in[3];
  const float* bq = (const float*)d_in[4];
  const float* Wv = (const float*)d_in[5];
  const float* bv = (const float*)d_in[6];
  float* out = (float*)d_out;

  // workspace layout (f16): K[16384][64] | Q[16384][64] | Vt[4][64][4096] | Wt[3][64][512]
  half_t* Kf = (half_t*)d_ws;
  half_t* Qf = Kf + (size_t)NROW * HDIM;
  half_t* Vt = Qf + (size_t)NROW * HDIM;
  half_t* Wt = Vt + (size_t)BATCH * HDIM * T;

  const int wtElems = 3 * HDIM * CDIM;
  prep_weights<<<(wtElems + 255) / 256, 256, 0, stream>>>(Wk, Wq, Wv, Wt);
  proj_kernel<<<NROW / 16, 32, 0, stream>>>(x, Wt, bk, bq, bv, Kf, Qf, Vt);
  attn_kernel<<<NROW / 16, 32, 0, stream>>>(Kf, Qf, Vt, out);
}
